// ThunderHubLM_46428596470332
// MI455X (gfx1250) — compile-verified, matrix operand order checked by reading, not run on hardware
//
#include <hip/hip_runtime.h>
#include <hip/hip_bf16.h>

#define D_ 256
#define S_ 8
#define DEPTH_ 4
#define B_ 2
#define L_ 2048
#define M_ (B_ * L_)        // 4096 rows
#define E_ (D_ + S_)        // 264
#define EPAD_ 288           // 264 padded to multiple of 32 (K for Wo GEMM)
#define NPAD1_ 320          // 264 padded to multiple of 64 (N for Wi GEMM)
#define VOCAB_ 32000

typedef __attribute__((ext_vector_type(16))) __bf16 v16bf;
typedef __attribute__((ext_vector_type(8)))  float  v8f;

union FragBF16 { uint4 q[2]; v16bf v; };

__device__ __forceinline__ unsigned short f32_to_bf16_rne(float f) {
    unsigned u = __float_as_uint(f);
    u += 0x7FFFu + ((u >> 16) & 1u);
    return (unsigned short)(u >> 16);
}

// ---------------------------------------------------------------------------
// Generic WMMA bf16 GEMM:  C[M x N] = A[M x K](bf16,row-major) * BT[N x K]^T
//                                     + bias[N]   (fp32 accumulate)
// block = 128 threads (4 waves). Wave w computes rows [row0+w*16, +16),
// cols [col0, col0+64) as 4 WMMA accumulators sharing one A fragment.
// K must be a multiple of 32; BT rows beyond N must be zero-padded.
// ---------------------------------------------------------------------------
__global__ void __launch_bounds__(128)
wmma_gemm_bf16_nt(const unsigned short* __restrict__ A, int lda,
                  const unsigned short* __restrict__ BT, int ldb,
                  const float* __restrict__ bias,
                  float* __restrict__ C, int ldc,
                  int M, int N, int K)
{
    const int wave  = threadIdx.x >> 5;
    const int lane  = threadIdx.x & 31;
    const int mlane = lane & 15;       // M (for A) / N (for B) within tile
    const int hi    = lane >> 4;       // K-half selector per ISA layout

    const int row0 = blockIdx.y * 64 + wave * 16;
    const int col0 = blockIdx.x * 64;

    v8f acc[4] = {};

    const unsigned short* arow = A + (size_t)(row0 + mlane) * lda + hi * 8;

    for (int k0 = 0; k0 < K; k0 += 32) {
        FragBF16 a;
        a.q[0] = *(const uint4*)(arow + k0);        // K = k0 + hi*8 .. +7
        a.q[1] = *(const uint4*)(arow + k0 + 16);   // K = k0+16+hi*8 .. +7
#pragma unroll
        for (int t = 0; t < 4; ++t) {
            const unsigned short* brow =
                BT + (size_t)(col0 + t * 16 + mlane) * ldb + hi * 8 + k0;
            FragBF16 b;
            b.q[0] = *(const uint4*)(brow);
            b.q[1] = *(const uint4*)(brow + 16);
            acc[t] = __builtin_amdgcn_wmma_f32_16x16x32_bf16(
                false, a.v, false, b.v, (short)0, acc[t], false, false);
        }
    }

#pragma unroll
    for (int t = 0; t < 4; ++t) {
        int n = col0 + t * 16 + mlane;
        if (n >= N) continue;
        float bs = bias ? bias[n] : 0.0f;
#pragma unroll
        for (int r = 0; r < 8; ++r) {
            int m = row0 + r + hi * 8;   // C/D layout: VGPR r, lanes>=16 -> M+8
            if (m < M) C[(size_t)m * ldc + n] = acc[t][r] + bs;
        }
    }
}

// ---------------------------------------------------------------------------
// Transpose + convert fp32 weight W[K x N] -> bf16 BT[Nrows x ldb], zero pad.
// ---------------------------------------------------------------------------
__global__ void transpose_to_bf16(const float* __restrict__ W, int K, int N,
                                  unsigned short* __restrict__ BT,
                                  int ldb, int Nrows)
{
    int idx = blockIdx.x * blockDim.x + threadIdx.x;
    int total = Nrows * ldb;
    if (idx >= total) return;
    int n = idx / ldb, k = idx % ldb;
    float v = (n < N && k < K) ? W[(size_t)k * N + n] : 0.0f;
    BT[idx] = f32_to_bf16_rne(v);
}

__global__ void cvt_f32_to_bf16(const float* __restrict__ s,
                                unsigned short* __restrict__ d, int n)
{
    int i = blockIdx.x * blockDim.x + threadIdx.x;
    if (i < n) d[i] = f32_to_bf16_rne(s[i]);
}

__global__ void embed_gather(const int* __restrict__ x,
                             const float* __restrict__ table,
                             float* __restrict__ h, int total)
{
    int i = blockIdx.x * blockDim.x + threadIdx.x;
    if (i >= total) return;
    int m = i >> 8, d = i & 255;
    h[i] = table[(size_t)x[m] * D_ + d];
}

// gate = sigmoid(combined @ Wg + bg), combined:[M x 264], Wg:[264 x 8]
__global__ void gate_kernel(const float* __restrict__ combined,
                            const float* __restrict__ Wg,
                            const float* __restrict__ bg,
                            float* __restrict__ gate, int total)
{
    int i = blockIdx.x * blockDim.x + threadIdx.x;
    if (i >= total) return;
    int m = i >> 3, s = i & 7;
    float acc = bg[s];
    const float* crow = combined + (size_t)m * E_;
#pragma unroll 4
    for (int e = 0; e < E_; ++e) acc = fmaf(crow[e], Wg[e * S_ + s], acc);
    gate[i] = 1.0f / (1.0f + __expf(-acc));
}

// 16 independent recurrences (b in 0..1, s in 0..7), sequential over L.
__global__ void scan_kernel(const float* __restrict__ combined,
                            const float* __restrict__ gate,
                            float* __restrict__ states,
                            float* __restrict__ out_final)
{
    int t = threadIdx.x;
    if (t >= B_ * S_) return;
    int b = t >> 3, s = t & 7;
    float st = 0.0f;
    for (int l = 0; l < L_; ++l) {
        int m = b * L_ + l;
        float g  = gate[m * S_ + s];
        float si = combined[(size_t)m * E_ + D_ + s];
        st = g * st + (1.0f - g) * si;
        states[m * S_ + s] = st;
    }
    out_final[b * S_ + s] = st;   // states output: (layer, B, S)
}

// concat[M x 288] bf16 = [combined[:, :256] | states[:, :8] | zeros(24)]
__global__ void build_concat(const float* __restrict__ combined,
                             const float* __restrict__ states,
                             unsigned short* __restrict__ out, int total)
{
    int i = blockIdx.x * blockDim.x + threadIdx.x;
    if (i >= total) return;
    int m = i / EPAD_, c = i % EPAD_;
    float v;
    if (c < D_)      v = combined[(size_t)m * E_ + c];
    else if (c < E_) v = states[m * S_ + (c - D_)];
    else             v = 0.0f;
    out[i] = f32_to_bf16_rne(v);
}

// LayerNorm over D=256 per row, output bf16 (feeds the two final GEMMs).
__global__ void __launch_bounds__(256)
layernorm_bf16(const float* __restrict__ h,
               const float* __restrict__ gamma,
               const float* __restrict__ beta,
               unsigned short* __restrict__ out)
{
    __shared__ float red[256];
    int t = threadIdx.x, m = blockIdx.x;
    float v = h[(size_t)m * D_ + t];
    red[t] = v; __syncthreads();
    for (int s = 128; s > 0; s >>= 1) {
        if (t < s) red[t] += red[t + s];
        __syncthreads();
    }
    float mu = red[0] * (1.0f / D_);
    __syncthreads();
    float d = v - mu;
    red[t] = d * d; __syncthreads();
    for (int s = 128; s > 0; s >>= 1) {
        if (t < s) red[t] += red[t + s];
        __syncthreads();
    }
    float var = red[0] * (1.0f / D_);
    float y = d * rsqrtf(var + 1e-5f) * gamma[t] + beta[t];
    out[(size_t)m * D_ + t] = f32_to_bf16_rne(y);
}

// ---------------------------------------------------------------------------
extern "C" void kernel_launch(void* const* d_in, const int* in_sizes, int n_in,
                              void* d_out, int out_size, void* d_ws, size_t ws_size,
                              hipStream_t stream)
{
    (void)in_sizes; (void)n_in; (void)out_size; (void)ws_size;

    const int*   x_t   = (const int*)  d_in[0];
    const float* table = (const float*)d_in[1];
    const float* Wi    = (const float*)d_in[2];   // (4, 256, 264)
    const float* bi    = (const float*)d_in[3];   // (4, 264)
    const float* Wg    = (const float*)d_in[4];   // (4, 264, 8)
    const float* bg    = (const float*)d_in[5];   // (4, 8)
    const float* Wo    = (const float*)d_in[6];   // (4, 264, 256)
    const float* bo    = (const float*)d_in[7];   // (4, 256)
    const float* gamma = (const float*)d_in[8];
    const float* beta  = (const float*)d_in[9];
    const float* Wc    = (const float*)d_in[10];  // (256, 32000)
    const float* bc    = (const float*)d_in[11];
    const float* Wr    = (const float*)d_in[12];  // (256, 256)
    const float* br    = (const float*)d_in[13];

    float* out_logits = (float*)d_out;                               // M x 32000
    float* out_recon  = out_logits + (size_t)M_ * VOCAB_;            // M x 256
    float* out_states = out_recon  + (size_t)M_ * D_;                // 4 x 2 x 8

    // ---- workspace partition (256B aligned slices) ----
    char* ws = (char*)d_ws;
    size_t off = 0;
    auto alloc = [&](size_t bytes) -> char* {
        char* p = ws + off;
        off += (bytes + 255) & ~size_t(255);
        return p;
    };
    float* h        = (float*)alloc((size_t)M_ * D_ * 4);
    float* combined = (float*)alloc((size_t)M_ * E_ * 4);
    float* gate     = (float*)alloc((size_t)M_ * S_ * 4);
    float* states   = (float*)alloc((size_t)M_ * S_ * 4);
    unsigned short* hA      = (unsigned short*)alloc((size_t)M_ * D_ * 2);
    unsigned short* concatA = (unsigned short*)alloc((size_t)M_ * EPAD_ * 2);
    unsigned short* WiT = (unsigned short*)alloc((size_t)DEPTH_ * NPAD1_ * D_ * 2);
    unsigned short* WoT = (unsigned short*)alloc((size_t)DEPTH_ * D_ * EPAD_ * 2);
    unsigned short* WcT = (unsigned short*)alloc((size_t)VOCAB_ * D_ * 2);
    unsigned short* WrT = (unsigned short*)alloc((size_t)D_ * D_ * 2);

    const int TPB = 256;
    auto blocks = [](size_t n, int tpb) { return (unsigned)((n + tpb - 1) / tpb); };

    // ---- weight prep: transpose to N-major bf16 (zero-padded) ----
    transpose_to_bf16<<<blocks((size_t)VOCAB_ * D_, TPB), TPB, 0, stream>>>(
        Wc, D_, VOCAB_, WcT, D_, VOCAB_);
    transpose_to_bf16<<<blocks((size_t)D_ * D_, TPB), TPB, 0, stream>>>(
        Wr, D_, D_, WrT, D_, D_);
    for (int lyr = 0; lyr < DEPTH_; ++lyr) {
        transpose_to_bf16<<<blocks((size_t)NPAD1_ * D_, TPB), TPB, 0, stream>>>(
            Wi + (size_t)lyr * D_ * E_, D_, E_,
            WiT + (size_t)lyr * NPAD1_ * D_, D_, NPAD1_);
        transpose_to_bf16<<<blocks((size_t)D_ * EPAD_, TPB), TPB, 0, stream>>>(
            Wo + (size_t)lyr * E_ * D_, E_, D_,
            WoT + (size_t)lyr * D_ * EPAD_, EPAD_, D_);
    }

    // ---- embedding ----
    embed_gather<<<blocks((size_t)M_ * D_, TPB), TPB, 0, stream>>>(
        x_t, table, h, M_ * D_);

    // ---- layers ----
    for (int lyr = 0; lyr < DEPTH_; ++lyr) {
        cvt_f32_to_bf16<<<blocks((size_t)M_ * D_, TPB), TPB, 0, stream>>>(
            h, hA, M_ * D_);

        // combined = h @ Wi + bi   (M=4096, N=264->5 tiles, K=256)
        wmma_gemm_bf16_nt<<<dim3(NPAD1_ / 64, M_ / 64), 128, 0, stream>>>(
            hA, D_, WiT + (size_t)lyr * NPAD1_ * D_, D_,
            bi + (size_t)lyr * E_, combined, E_, M_, E_, D_);

        gate_kernel<<<blocks((size_t)M_ * S_, TPB), TPB, 0, stream>>>(
            combined, Wg + (size_t)lyr * E_ * S_, bg + (size_t)lyr * S_,
            gate, M_ * S_);

        scan_kernel<<<1, 32, 0, stream>>>(
            combined, gate, states, out_states + (size_t)lyr * B_ * S_);

        build_concat<<<blocks((size_t)M_ * EPAD_, TPB), TPB, 0, stream>>>(
            combined, states, concatA, M_ * EPAD_);

        // h = concat @ Wo + bo   (M=4096, N=256, K=288)
        wmma_gemm_bf16_nt<<<dim3(D_ / 64, M_ / 64), 128, 0, stream>>>(
            concatA, EPAD_, WoT + (size_t)lyr * D_ * EPAD_, EPAD_,
            bo + (size_t)lyr * D_, h, D_, M_, D_, EPAD_);
    }

    // ---- layernorm -> bf16 activations ----
    layernorm_bf16<<<M_, 256, 0, stream>>>(h, gamma, beta, hA);

    // ---- logits = hn @ Wc + bc  (M=4096, N=32000, K=256) : store-bound ----
    wmma_gemm_bf16_nt<<<dim3(VOCAB_ / 64, M_ / 64), 128, 0, stream>>>(
        hA, D_, WcT, D_, bc, out_logits, VOCAB_, M_, VOCAB_, D_);

    // ---- recon = hn @ Wr + br ----
    wmma_gemm_bf16_nt<<<dim3(D_ / 64, M_ / 64), 128, 0, stream>>>(
        hA, D_, WrT, D_, br, out_recon, D_, M_, D_, D_);
}
